// STDPLinear_6347961664040
// MI455X (gfx1250) — matrix operand order repaired and checked by reading, not compiled, not run on hardware
//
#include <hip/hip_runtime.h>
#include <hip/hip_bf16.h>

typedef __attribute__((ext_vector_type(16))) _Float16 v16h;
typedef __attribute__((ext_vector_type(8)))  _Float16 v8h;
typedef __attribute__((ext_vector_type(8)))  float    v8f;

#define TSTEPS 2048
#define NFEAT  2048
#define A_POS   0.05f
#define TAU_POS 2.0f
#define A_NEG   0.05f
#define TAU_NEG 2.0f
#define LR      0.01f

// ---------------------------------------------------------------------------
// Build STDP kernel matrix K[ti, tj] = f(ti - tj) in f16, row-major [T, T].
// In f16, K rounds to exactly zero for |ti - tj| >= 29 (0.05*e^{-dt/2} < 2^-25),
// which the banded GEMM1 below exploits bit-exactly.
// ---------------------------------------------------------------------------
__global__ __launch_bounds__(256) void build_K(_Float16* __restrict__ Kh) {
    int idx  = blockIdx.x * blockDim.x + threadIdx.x;   // one thread per 8 elems
    int row  = idx >> 8;                                // T/8 = 256 groups per row
    int col0 = (idx & 255) << 3;
    if (row >= TSTEPS) return;
    v8h out;
#pragma unroll
    for (int e = 0; e < 8; ++e) {
        float dt = (float)(row - (col0 + e));
        float k  = (dt > 0.0f) ? (A_POS * __expf(-dt / TAU_POS))
                               : (-A_NEG * __expf(dt / TAU_NEG));
        out[e] = (_Float16)k;
    }
    *(v8h*)(Kh + (long)row * TSTEPS + col0) = out;
}

// ---------------------------------------------------------------------------
// LDS-tiled transpose with f32 -> f16 convert: dst[c*R + r] = src[r*C + c]
// ---------------------------------------------------------------------------
__global__ __launch_bounds__(256) void transpose_to_f16(const float* __restrict__ src,
                                                        _Float16* __restrict__ dst,
                                                        int R, int C) {
    __shared__ float tile[32][33];
    int bx = blockIdx.x * 32;           // col base in src
    int by = blockIdx.y * 32;           // row base in src
    int tx = threadIdx.x & 31;
    int ty = threadIdx.x >> 5;          // 0..7
#pragma unroll
    for (int i = ty; i < 32; i += 8)
        tile[i][tx] = src[(long)(by + i) * C + (bx + tx)];
    __syncthreads();
#pragma unroll
    for (int i = ty; i < 32; i += 8)
        dst[(long)(bx + i) * R + (by + tx)] = (_Float16)tile[tx][i];
}

// ---------------------------------------------------------------------------
// WMMA fragment loads (wave32), same contiguous k order (k = 16*chunk + e) in A and B.
// A (16x32 f16, row-major, lda elems): lane = M%16 + 16*chunk;
//   v16h elems 0..7  <- A[m, chunk*16 + 0..7]
//   v16h elems 8..15 <- A[m, chunk*16 + 8 + 0..7]
// B (32x16 f16) supplied as Bt row-major [N, K]: lane n = L%16, kb = (L/16)*16;
//   v16h elem e <- B[kb+e, n] = Bt[n, kb+e]  (16 contiguous halves)
// ---------------------------------------------------------------------------
__device__ __forceinline__ v16h load_fragA(const _Float16* __restrict__ A,
                                           int lda, int lane) {
    int m  = lane & 15;
    int ch = lane >> 4;
    const _Float16* p = A + (long)m * lda + ch * 16;
    v8h lo = *(const v8h*)p;
    v8h hi = *(const v8h*)(p + 8);
    v16h r;
#pragma unroll
    for (int e = 0; e < 8; ++e) { r[e] = lo[e]; r[e + 8] = hi[e]; }
    return r;
}

__device__ __forceinline__ v16h load_fragB(const _Float16* __restrict__ Bt,
                                           int ldb, int lane) {
    int n  = lane & 15;
    int kb = (lane >> 4) << 4;
    return *(const v16h*)(Bt + (long)n * ldb + kb);
}

// ---------------------------------------------------------------------------
// NT GEMM with transposed (contiguous) store.
//   C[m,n] = sum_k A[m,k] * Bt[n,k]
// Wave tile 32x64: 2 A-frags x 4 B-frags -> 8 WMMAs per 192B/lane of loads.
// Block = 4 waves -> 64x128 C tile.
// BANDED==1 (GEMM1, A = f16 STDP Toeplitz matrix): only k-tiles within the
//   exact f16 nonzero band |m-k| <= 28 are visited ([mBase-32, mBase+64)),
//   bit-identical to the full loop since K underflows to 0 outside it.
// FINAL==0: Ct[n*M + m] = (f16)C
// FINAL==1: Out[n*M + m] = W[n*M + m] + LR * C   (f32 fused epilogue)
// D/C layout: lane L -> n = L%16 ; VGPR g -> m = 8*(L/16) + g, so each lane
// stores 8 consecutive m at one C^T row: fully contiguous 16B/32B stores.
// ---------------------------------------------------------------------------
template <int FINAL, int BANDED>
__global__ __launch_bounds__(128) void gemm_nt(const _Float16* __restrict__ A,
                                               const _Float16* __restrict__ Bt,
                                               _Float16* __restrict__ Ct,
                                               const float* __restrict__ W,
                                               float* __restrict__ Out,
                                               int M, int N, int Kdim) {
    const int lane  = threadIdx.x & 31;
    const int wave  = threadIdx.x >> 5;                 // 0..3
    const int mBase = blockIdx.x * 64  + (wave & 1) * 32;
    const int nBase = blockIdx.y * 128 + (wave >> 1) * 64;

    int kLo = 0, kHi = Kdim;
    if (BANDED) {
        kLo = mBase >= 32 ? mBase - 32 : 0;
        kHi = (mBase + 64) < Kdim ? (mBase + 64) : Kdim;
    }

    v8f acc[2][4] = {};
    for (int k = kLo; k < kHi; k += 32) {
        v16h a[2], b[4];
#pragma unroll
        for (int mi = 0; mi < 2; ++mi)
            a[mi] = load_fragA(A + (long)(mBase + 16 * mi) * Kdim + k, Kdim, lane);
#pragma unroll
        for (int ni = 0; ni < 4; ++ni)
            b[ni] = load_fragB(Bt + (long)(nBase + 16 * ni) * Kdim + k, Kdim, lane);
#pragma unroll
        for (int mi = 0; mi < 2; ++mi)
#pragma unroll
            for (int ni = 0; ni < 4; ++ni)
                acc[mi][ni] = __builtin_amdgcn_wmma_f32_16x16x32_f16(
                    false, a[mi], false, b[ni], (short)0, acc[mi][ni], false, false);
    }

    const int nLoc = lane & 15;
    const int mOff = (lane >> 4) * 8;
#pragma unroll
    for (int mi = 0; mi < 2; ++mi) {
#pragma unroll
        for (int ni = 0; ni < 4; ++ni) {
            long row  = (long)(nBase + ni * 16 + nLoc);   // C^T row index (n)
            long colb = (long)(mBase + mi * 16 + mOff);   // 8 consecutive m
            if (FINAL) {
                const float* w = W + row * M + colb;
                float*       o = Out + row * M + colb;
                v8f r;
#pragma unroll
                for (int g = 0; g < 8; ++g) r[g] = w[g] + LR * acc[mi][ni][g];
                *(v8f*)o = r;
            } else {
                v8h h;
#pragma unroll
                for (int g = 0; g < 8; ++g) h[g] = (_Float16)acc[mi][ni][g];
                *(v8h*)(Ct + row * M + colb) = h;
            }
        }
    }
}

// ---------------------------------------------------------------------------
// Pipeline:
//   Kh   = f16 STDP Toeplitz matrix [T, T]   (banded: zero for |dt| >= 29)
//   inT  = f16 transpose of input spikes  [I, T]
//   outT = f16 transpose of output spikes [O, T]
//   GEMM1 (NT, banded, t-store): kinT[i,t] = sum_tj Kh[t,tj] * inT[i,tj]
//   GEMM2 (NT, t-store):         Out[o,i]  = W[o,i] + LR*sum_t kinT[i,t]*outT[o,t]
// ---------------------------------------------------------------------------
extern "C" void kernel_launch(void* const* d_in, const int* in_sizes, int n_in,
                              void* d_out, int out_size, void* d_ws, size_t ws_size,
                              hipStream_t stream) {
    (void)in_sizes; (void)n_in; (void)out_size; (void)ws_size;

    const float* weight  = (const float*)d_in[0];
    const float* in_spk  = (const float*)d_in[1];
    const float* out_spk = (const float*)d_in[2];
    float*       out     = (float*)d_out;

    char* ws = (char*)d_ws;
    const size_t mat_h = (size_t)TSTEPS * NFEAT * sizeof(_Float16);  // 8 MB
    _Float16* Kh   = (_Float16*)(ws);
    _Float16* inT  = (_Float16*)(ws + 1 * mat_h);
    _Float16* outT = (_Float16*)(ws + 2 * mat_h);
    _Float16* kinT = (_Float16*)(ws + 3 * mat_h);

    // 1) STDP kernel matrix
    build_K<<<(TSTEPS * TSTEPS / 8) / 256, 256, 0, stream>>>(Kh);

    // 2) Transpose + f16 convert of both spike trains
    dim3 tb(256), tg(NFEAT / 32, TSTEPS / 32);
    transpose_to_f16<<<tg, tb, 0, stream>>>(in_spk,  inT,  TSTEPS, NFEAT);
    transpose_to_f16<<<tg, tb, 0, stream>>>(out_spk, outT, TSTEPS, NFEAT);

    // 3) GEMM1 (banded): kinT = (K @ in)^T   [I, T] f16
    dim3 gb(128), gg1(TSTEPS / 64, NFEAT / 128);
    gemm_nt<0, 1><<<gg1, gb, 0, stream>>>(Kh, inT, kinT, nullptr, nullptr,
                                          TSTEPS, NFEAT, TSTEPS);

    // 4) GEMM2 fused epilogue: Out = W + LR * out^T @ kin   [O, I] f32
    dim3 gg2(NFEAT / 64, NFEAT / 128);
    gemm_nt<1, 0><<<gg2, gb, 0, stream>>>(kinT, outT, nullptr, weight, out,
                                          NFEAT, NFEAT, TSTEPS);
}
